// Attention_27805618275053
// MI455X (gfx1250) — matrix operand order repaired and checked by reading, not compiled, not run on hardware
//
#include <hip/hip_runtime.h>

// ---------------------------------------------------------------------------
// Problem constants (match the JAX reference)
// ---------------------------------------------------------------------------
#define B_   256
#define N_   196          // tokens per window (14x14)
#define C_   384
#define H_   8
#define KD_  32
#define VD_  128
#define QKVD 192          // per-head qkv width (32+32+128)
#define M_   (B_ * N_)    // 50176 rows
#define NQKV (H_ * QKVD)  // 1536
#define NOUT (H_ * VD_)   // 1024

#define SCALE_F 0.17677669529663687f   // 32^-0.5

// attention tiling
#define QPAD 208          // 13 * 16 query rows
#define NPAD 224          // 7 * 32 key cols (for K=32 WMMA chunks)
#define QS_STRIDE 40
#define KS_STRIDE 40
#define VT_STRIDE 232
#define PS_STRIDE 232

typedef __bf16  v16bf __attribute__((ext_vector_type(16)));
typedef float   v8f   __attribute__((ext_vector_type(8)));
typedef unsigned int u32x4 __attribute__((ext_vector_type(4)));
typedef int          i32x4 __attribute__((ext_vector_type(4)));
typedef int          i32x8 __attribute__((ext_vector_type(8)));

// Tensor Data Mover availability (this toolchain: 6-arg builtin,
// amdgpu-toolchain / clang-23 form)
#if defined(__has_builtin)
#if __has_builtin(__builtin_amdgcn_tensor_load_to_lds) && \
    __has_builtin(__builtin_amdgcn_s_wait_tensorcnt)
#define HAVE_TDM 1
#endif
#endif
#ifndef HAVE_TDM
#define HAVE_TDM 0
#endif

// ---------------------------------------------------------------------------
// Fragment helpers for V_WMMA_F32_16X16X32_BF16 (wave32; A/B use contiguous k, not the ISA table order):
//  A (16xK=32):  lane L -> row M = L%16 ; half g=L/16 :
//                vgpr j(0..7)  holds K = g*16 + 2j, 2j+1
//  B (K=32x16):  lane L -> col N = L%16 ; half g=L/16 :
//                vgpr j holds K = g*16 + 2j, 2j+1
//  C/D (16x16):  vgpr j -> row M = j + 8*(L/16) ; col N = L%16
// ---------------------------------------------------------------------------
__device__ __forceinline__ v16bf load_a_frag(const __bf16* row, int half) {
  v16bf a;
#pragma unroll
  for (int j = 0; j < 4; ++j) {
    int k = half * 16 + 2 * j;
    a[2 * j] = row[k];  a[2 * j + 1] = row[k + 1];
  }
#pragma unroll
  for (int j = 0; j < 4; ++j) {
    int k = half * 16 + 8 + 2 * j;
    a[8 + 2 * j] = row[k];  a[8 + 2 * j + 1] = row[k + 1];
  }
  return a;
}

__device__ __forceinline__ v16bf load_b_frag(const __bf16* colK0) {
  // colK0 points at (K = half*16, N = lane%16) with K contiguous
  v16bf b;
#pragma unroll
  for (int j = 0; j < 8; ++j) {
    b[2 * j] = colK0[2 * j];  b[2 * j + 1] = colK0[2 * j + 1];
  }
  return b;
}

#if HAVE_TDM
// ---------------------------------------------------------------------------
// TDM: async 2D tile load Global->LDS with hardware row padding.
// tile_w elements of data_size=2B per row, tile_h rows, global row stride
// row_stride elements. LDS gets 16B of pad after every 64B row
// (pad_interval code 3 = 16 DWORDs, pad_amount code 3 = 4 DWORDs) so the
// LDS row stride is exactly 40 bf16 = QS_STRIDE/KS_STRIDE.
// D# layout per CDNA5 ISA ch.8 (group0 128b, group1 256b).
// ---------------------------------------------------------------------------
__device__ __forceinline__ void tdm_load_2d_pad(const void* gptr,
                                                unsigned lds_off,
                                                unsigned tile_w,
                                                unsigned tile_h,
                                                unsigned row_stride) {
  const unsigned long long ga = (unsigned long long)(uintptr_t)gptr;
  u32x4 g0;
  g0[0] = 1u;                                   // count=1 (valid), user mode
  g0[1] = lds_off;                              // lds_addr (bytes)
  g0[2] = (unsigned)(ga & 0xffffffffu);         // global_addr[31:0]
  g0[3] = (unsigned)((ga >> 32) & 0x01ffffffu)  // global_addr[56:32]
          | (2u << 30);                         // type=2 ("image")
  i32x8 g1;
  g1[0] = (int)((1u << 16)      // data_size = 1 -> 2 bytes
                | (1u << 20)    // pad_enable
                | (3u << 22)    // pad_interval: 16 DWORDs (64B) before pad
                | (3u << 25));  // pad_amount: 4 DWORDs (16B)
  g1[1] = (int)(tile_w << 16);  // tensor_dim0[15:0] (bits 63:48)
  g1[2] = (int)(tile_h << 16);  // tensor_dim1[15:0] (bits 95:80)
  g1[3] = (int)(tile_w << 16);  // tile_dim0 (bits 127:112)
  g1[4] = (int)(tile_h);        // tile_dim1 (bits 143:128) ; tile_dim2 = 0
  g1[5] = (int)(row_stride);    // tensor_dim0_stride[31:0] (bits 191:160)
  g1[6] = 0;                    // stride hi, tensor_dim1_stride lo
  g1[7] = 0;
  const i32x4 z4 = {0, 0, 0, 0};          // groups 2/3 unused (2D tensor)
  const i32x8 z8 = {0, 0, 0, 0, 0, 0, 0, 0};
  __builtin_amdgcn_tensor_load_to_lds(g0, g1, z4, z4, z8, 0);
}
#endif

// ---------------------------------------------------------------------------
// K0: convert both weight matrices to bf16 (concatenated in workspace)
// ---------------------------------------------------------------------------
__global__ void __launch_bounds__(256)
cvt_weights(const float* __restrict__ wqkv, const float* __restrict__ wproj,
            __bf16* __restrict__ dst) {
  const int NQ = NQKV * C_;        // 589824
  const int NP = C_ * NOUT;        // 393216
  int i = blockIdx.x * 256 + threadIdx.x;
  if (i < NQ)            dst[i] = (__bf16)wqkv[i];
  else if (i < NQ + NP)  dst[i] = (__bf16)wproj[i - NQ];
}

// ---------------------------------------------------------------------------
// K1: qkv = x @ w_qkv^T + b_qkv   (f32 in, bf16 out, (B,H,N,192) layout)
//     block tile 128x64, k-step 32, 8 waves as 4x2 of 32x32
// ---------------------------------------------------------------------------
__global__ void __launch_bounds__(256)
qkv_gemm(const float* __restrict__ x,      // (M_, 384)
         const __bf16* __restrict__ w,     // (1536, 384) bf16
         const float* __restrict__ bias,   // (1536)
         __bf16* __restrict__ qkv) {       // (B,H,196,192)
  __shared__ __bf16 As[128][40];
  __shared__ __bf16 Bs[64][40];

  const int mt = blockIdx.x % (M_ / 128);      // 392 tiles
  const int nt = blockIdx.x / (M_ / 128);      // 24 tiles
  const int m0 = mt * 128, n0 = nt * 64;
  const int tid  = threadIdx.x;
  const int lane = tid & 31, wave = tid >> 5;
  const int wm = wave & 3, wn = wave >> 2;
  const int half = lane >> 4, l16 = lane & 15;

  v8f acc[2][2];
#pragma unroll
  for (int i = 0; i < 2; ++i)
#pragma unroll
    for (int j = 0; j < 2; ++j) acc[i][j] = {};

  for (int ks = 0; ks < C_ / 32; ++ks) {
    const int k0 = ks * 32;
    __syncthreads();
    // A tile: 128x32 f32 -> bf16 LDS
    {
      const int colg  = (tid & 7) * 4;
      const int rbase = tid >> 3;
#pragma unroll
      for (int i = 0; i < 4; ++i) {
        const int r = rbase + 32 * i;
        const float4 f = *(const float4*)(x + (size_t)(m0 + r) * C_ + k0 + colg);
        __bf16* d = &As[r][colg];
        d[0] = (__bf16)f.x; d[1] = (__bf16)f.y; d[2] = (__bf16)f.z; d[3] = (__bf16)f.w;
      }
    }
    // B tile: 64x32 bf16
    {
      const int r = tid >> 2, cg = (tid & 3) * 8;
      *(uint4*)&Bs[r][cg] = *(const uint4*)(w + (size_t)(n0 + r) * C_ + k0 + cg);
    }
    __syncthreads();
    if (ks < C_ / 32 - 1) {
      // gfx1250 global_prefetch of next A k-slice
      __builtin_prefetch(x + (size_t)(m0 + (tid >> 3)) * C_ + k0 + 32, 0, 1);
    }
    v16bf a[2], b[2];
#pragma unroll
    for (int i = 0; i < 2; ++i)
      a[i] = load_a_frag(&As[wm * 32 + i * 16 + l16][0], half);
#pragma unroll
    for (int i = 0; i < 2; ++i)
      b[i] = load_b_frag(&Bs[wn * 32 + i * 16 + l16][half * 16]);
#pragma unroll
    for (int i = 0; i < 2; ++i)
#pragma unroll
      for (int jj = 0; jj < 2; ++jj)
        acc[i][jj] = __builtin_amdgcn_wmma_f32_16x16x32_bf16(
            false, a[i], false, b[jj], (short)0, acc[i][jj], false, false);
  }

  // epilogue: +bias, scatter into (B,H,N,192) bf16
#pragma unroll
  for (int jj = 0; jj < 2; ++jj) {
    const int c  = n0 + wn * 32 + jj * 16 + l16;
    const float bb = bias[c];
    const int h = c / QKVD, d = c % QKVD;
#pragma unroll
    for (int i = 0; i < 2; ++i) {
#pragma unroll
      for (int j = 0; j < 8; ++j) {
        const int m = m0 + wm * 32 + i * 16 + half * 8 + j;
        const int b_ = m / N_, n = m % N_;
        const float v = acc[i][jj][j] + bb;
        qkv[(((size_t)(b_ * H_ + h) * N_ + n)) * QKVD + d] = (__bf16)v;
      }
    }
  }
}

// ---------------------------------------------------------------------------
// K2: fused attention per (b,h): S = QK^T*scale + bias, softmax, O = P V
//     one workgroup per (b,h); 8 waves each own 16-row query tiles.
//     Q/K tiles staged by the Tensor Data Mover (hardware-padded rows);
//     V staged transposed by VALU path.
// ---------------------------------------------------------------------------
__global__ void __launch_bounds__(256)
attn_kernel(const __bf16* __restrict__ qkv,        // (B,H,196,192)
            const float* __restrict__ attn_biases, // (8,196)
            const int*   __restrict__ bias_idxs,   // (196,196)
            __bf16* __restrict__ O) {              // (B,196,1024)
  extern __shared__ char smem[];
  __bf16* Qs = (__bf16*)smem;                        // [208][40]
  __bf16* Ks = Qs + QPAD * QS_STRIDE;                // [224][40]
  __bf16* Vt = Ks + NPAD * KS_STRIDE;                // [128][232] (transposed V)
  __bf16* Ps = Vt + VD_ * VT_STRIDE;                 // [8][16][232]
  float*  biasrow = (float*)(Ps + 8 * 16 * PS_STRIDE); // [196]

  const int bh = blockIdx.x;
  const int b = bh / H_, h = bh % H_;
  const int tid = threadIdx.x;
  const int lane = tid & 31, wave = tid >> 5;
  const int half = lane >> 4, l16 = lane & 15;
  const __bf16* src = qkv + (size_t)(b * H_ + h) * N_ * QKVD;

#if HAVE_TDM
  // Q (196x32) and K (196x32) tiles: global row stride 192 elems, LDS rows
  // padded 64B->80B by the TDM so they land at stride QS_STRIDE/KS_STRIDE.
  if (wave == 0) {
    tdm_load_2d_pad(src,       (unsigned)(uintptr_t)Qs, KD_, N_, QKVD);
    tdm_load_2d_pad(src + KD_, (unsigned)(uintptr_t)Ks, KD_, N_, QKVD);
  }
#endif

  // stage this head's bias table row (196 floats)
  for (int i = tid; i < N_; i += 256) biasrow[i] = attn_biases[h * N_ + i];

#if HAVE_TDM
  // cooperative load of V only (transposed into Vt)
  for (int c = tid; c < N_ * 32; c += 256) {
    const int row = c / 32, dd = (c % 32) * 4;
    const __bf16* p = src + (size_t)row * QKVD + 2 * KD_ + dd;
    Vt[(dd + 0) * VT_STRIDE + row] = p[0];
    Vt[(dd + 1) * VT_STRIDE + row] = p[1];
    Vt[(dd + 2) * VT_STRIDE + row] = p[2];
    Vt[(dd + 3) * VT_STRIDE + row] = p[3];
  }
#else
  // cooperative load: each 4-bf16 chunk routed to Qs / Ks / Vt (V transposed)
  for (int c = tid; c < N_ * 48; c += 256) {
    const int row = c / 48, d = (c % 48) * 4;
    const __bf16* p = src + (size_t)row * QKVD + d;
    const __bf16 x0 = p[0], x1 = p[1], x2 = p[2], x3 = p[3];
    if (d < KD_) {
      __bf16* q = Qs + row * QS_STRIDE + d;
      q[0] = x0; q[1] = x1; q[2] = x2; q[3] = x3;
    } else if (d < 2 * KD_) {
      __bf16* k = Ks + row * KS_STRIDE + (d - KD_);
      k[0] = x0; k[1] = x1; k[2] = x2; k[3] = x3;
    } else {
      const int dd = d - 2 * KD_;
      Vt[(dd + 0) * VT_STRIDE + row] = x0;
      Vt[(dd + 1) * VT_STRIDE + row] = x1;
      Vt[(dd + 2) * VT_STRIDE + row] = x2;
      Vt[(dd + 3) * VT_STRIDE + row] = x3;
    }
  }
#endif
  // zero padding (so WMMA on pad keys/queries stays finite)
  for (int r = N_ + tid; r < QPAD; r += 256)
    for (int d = 0; d < KD_; ++d) Qs[r * QS_STRIDE + d] = (__bf16)0.f;
  for (int r = N_ + tid; r < NPAD; r += 256)
    for (int d = 0; d < KD_; ++d) Ks[r * KS_STRIDE + d] = (__bf16)0.f;
  for (int i = tid; i < VD_ * (NPAD - N_); i += 256) {
    const int r = i / (NPAD - N_), c = N_ + i % (NPAD - N_);
    Vt[r * VT_STRIDE + c] = (__bf16)0.f;
  }
#if HAVE_TDM
  __builtin_amdgcn_s_wait_tensorcnt(0);   // drain TDM before the barrier
#endif
  __syncthreads();

  __bf16* pw = Ps + wave * 16 * PS_STRIDE;

  for (int rt = wave; rt < QPAD / 16; rt += 8) {   // 13 row tiles over 8 waves
    // ---- S = Q K^T : one 16x16x32 WMMA per 16-key column tile ----
    const v16bf aq = load_a_frag(Qs + (rt * 16 + l16) * QS_STRIDE, half);
    v8f s[NPAD / 16];
#pragma unroll
    for (int ct = 0; ct < NPAD / 16; ++ct) {
      const v16bf bk = load_b_frag(Ks + (ct * 16 + l16) * KS_STRIDE + half * 16);
      v8f c = {};
      s[ct] = __builtin_amdgcn_wmma_f32_16x16x32_bf16(
          false, aq, false, bk, (short)0, c, false, false);
    }

    // ---- bias + masked softmax (row = j + 8*half within tile) ----
    const int rowbase = rt * 16 + half * 8;
    float rsum[8];
#pragma unroll
    for (int j = 0; j < 8; ++j) {
      const int row = rowbase + j;
      float m = -3.0e38f;
#pragma unroll
      for (int ct = 0; ct < NPAD / 16; ++ct) {
        const int col = ct * 16 + l16;
        float v = s[ct][j] * SCALE_F;
        if (row < N_ && col < N_) {
          const int idx = bias_idxs[row * N_ + col];
          v += biasrow[idx];
        } else {
          v = -1.0e30f;
        }
        s[ct][j] = v;
        m = fmaxf(m, v);
      }
#pragma unroll
      for (int off = 1; off < 16; off <<= 1) m = fmaxf(m, __shfl_xor(m, off));
      float sum = 0.f;
#pragma unroll
      for (int ct = 0; ct < NPAD / 16; ++ct) {
        const float e = __expf(s[ct][j] - m);
        s[ct][j] = e;
        sum += e;
      }
#pragma unroll
      for (int off = 1; off < 16; off <<= 1) sum += __shfl_xor(sum, off);
      rsum[j] = sum;
    }

    // ---- relayout P (C-layout regs) -> per-wave LDS (row-major bf16) ----
#pragma unroll
    for (int ct = 0; ct < NPAD / 16; ++ct)
#pragma unroll
      for (int j = 0; j < 8; ++j)
        pw[(half * 8 + j) * PS_STRIDE + ct * 16 + l16] = (__bf16)s[ct][j];
    // same-wave LDS RAW: drain DScnt (CDNA5 split wait counters)
    asm volatile("s_wait_dscnt 0" ::: "memory");

    // ---- O = P V : 8 vd-tiles x 7 key-chunks of K=32 ----
    v8f o[VD_ / 16];
#pragma unroll
    for (int vt = 0; vt < VD_ / 16; ++vt) o[vt] = {};
#pragma unroll
    for (int kc = 0; kc < NPAD / 32; ++kc) {
      const v16bf ap = load_a_frag(pw + l16 * PS_STRIDE + kc * 32, half);
#pragma unroll
      for (int vt = 0; vt < VD_ / 16; ++vt) {
        const v16bf bv =
            load_b_frag(Vt + (vt * 16 + l16) * VT_STRIDE + kc * 32 + half * 16);
        o[vt] = __builtin_amdgcn_wmma_f32_16x16x32_bf16(
            false, ap, false, bv, (short)0, o[vt], false, false);
      }
    }

    // ---- normalize + store bf16 O (B, N, H*VD) ----
#pragma unroll
    for (int vt = 0; vt < VD_ / 16; ++vt) {
#pragma unroll
      for (int j = 0; j < 8; ++j) {
        const int row = rowbase + j;
        if (row < N_) {
          const float val = o[vt][j] / rsum[j];
          O[(size_t)(b * N_ + row) * NOUT + h * VD_ + vt * 16 + l16] = (__bf16)val;
        }
      }
    }
  }
}

// ---------------------------------------------------------------------------
// K3: out = O @ w_proj^T + b_proj  (bf16 in, f32 out)
// ---------------------------------------------------------------------------
__global__ void __launch_bounds__(256)
proj_gemm(const __bf16* __restrict__ A,    // (M_, 1024)
          const __bf16* __restrict__ w,    // (384, 1024)
          const float* __restrict__ bias,  // (384)
          float* __restrict__ out) {       // (M_, 384)
  __shared__ __bf16 As[128][40];
  __shared__ __bf16 Bs[64][40];

  const int mt = blockIdx.x % (M_ / 128);
  const int nt = blockIdx.x / (M_ / 128);     // 6 tiles
  const int m0 = mt * 128, n0 = nt * 64;
  const int tid  = threadIdx.x;
  const int lane = tid & 31, wave = tid >> 5;
  const int wm = wave & 3, wn = wave >> 2;
  const int half = lane >> 4, l16 = lane & 15;

  v8f acc[2][2];
#pragma unroll
  for (int i = 0; i < 2; ++i)
#pragma unroll
    for (int j = 0; j < 2; ++j) acc[i][j] = {};

  for (int ks = 0; ks < NOUT / 32; ++ks) {
    const int k0 = ks * 32;
    __syncthreads();
#pragma unroll
    for (int i = 0; i < 2; ++i) {
      const int ch = tid + i * 256;           // 512 uint4 chunks
      const int r = ch >> 2, cg = (ch & 3) * 8;
      *(uint4*)&As[r][cg] = *(const uint4*)(A + (size_t)(m0 + r) * NOUT + k0 + cg);
    }
    {
      const int r = tid >> 2, cg = (tid & 3) * 8;
      *(uint4*)&Bs[r][cg] = *(const uint4*)(w + (size_t)(n0 + r) * NOUT + k0 + cg);
    }
    __syncthreads();
    if (ks < NOUT / 32 - 1)
      __builtin_prefetch(A + (size_t)(m0 + (tid >> 2)) * NOUT + k0 + 32, 0, 1);

    v16bf a[2], b[2];
#pragma unroll
    for (int i = 0; i < 2; ++i)
      a[i] = load_a_frag(&As[wm * 32 + i * 16 + l16][0], half);
#pragma unroll
    for (int i = 0; i < 2; ++i)
      b[i] = load_b_frag(&Bs[wn * 32 + i * 16 + l16][half * 16]);
#pragma unroll
    for (int i = 0; i < 2; ++i)
#pragma unroll
      for (int jj = 0; jj < 2; ++jj)
        acc[i][jj] = __builtin_amdgcn_wmma_f32_16x16x32_bf16(
            false, a[i], false, b[jj], (short)0, acc[i][jj], false, false);
  }

#pragma unroll
  for (int jj = 0; jj < 2; ++jj) {
    const int c = n0 + wn * 32 + jj * 16 + l16;
    const float bb = bias[c];
#pragma unroll
    for (int i = 0; i < 2; ++i)
#pragma unroll
      for (int j = 0; j < 8; ++j) {
        const int m = m0 + wm * 32 + i * 16 + half * 8 + j;
        out[(size_t)m * C_ + c] = acc[i][jj][j] + bb;
      }
  }
}

// ---------------------------------------------------------------------------
// Host launcher
// ---------------------------------------------------------------------------
extern "C" void kernel_launch(void* const* d_in, const int* in_sizes, int n_in,
                              void* d_out, int out_size, void* d_ws, size_t ws_size,
                              hipStream_t stream) {
  const float* x           = (const float*)d_in[0];
  const float* w_qkv       = (const float*)d_in[1];
  const float* b_qkv       = (const float*)d_in[2];
  const float* w_proj      = (const float*)d_in[3];
  const float* b_proj      = (const float*)d_in[4];
  const float* attn_biases = (const float*)d_in[5];
  const int*   bias_idxs   = (const int*)d_in[6];
  float* out = (float*)d_out;

  // workspace layout (bf16)
  __bf16* wqkv_b  = (__bf16*)d_ws;                         // 1536*384
  __bf16* wproj_b = wqkv_b + (size_t)NQKV * C_;            // 384*1024
  __bf16* qkv     = wproj_b + (size_t)C_ * NOUT;           // (B,H,196,192)
  __bf16* Obuf    = qkv + (size_t)M_ * NQKV;               // (B,196,1024)

  // K0: weight conversion (983040 elements)
  cvt_weights<<<(NQKV * C_ + C_ * NOUT + 255) / 256, 256, 0, stream>>>(
      w_qkv, w_proj, wqkv_b);

  // K1: qkv GEMM — 392 x 24 tiles
  qkv_gemm<<<(M_ / 128) * (NQKV / 64), 256, 0, stream>>>(x, wqkv_b, b_qkv, qkv);

  // K2: fused attention — one WG per (b,h); ~150 KB dynamic LDS (2 WGs / WGP)
  const size_t smem =
      (size_t)(QPAD * QS_STRIDE + NPAD * KS_STRIDE + VD_ * VT_STRIDE +
               8 * 16 * PS_STRIDE) * sizeof(__bf16) + N_ * sizeof(float);
  (void)hipFuncSetAttribute((const void*)attn_kernel,
                            hipFuncAttributeMaxDynamicSharedMemorySize,
                            (int)smem);
  attn_kernel<<<B_ * H_, 256, smem, stream>>>(qkv, attn_biases, bias_idxs, Obuf);

  // K3: output projection — 392 x 6 tiles
  proj_gemm<<<(M_ / 128) * (C_ / 64), 256, 0, stream>>>(Obuf, wproj_b, b_proj, out);
}
